// CausalSelfAttention_31284541784390
// MI455X (gfx1250) — matrix operand order repaired and checked by reading, not compiled, not run on hardware
//
#include <hip/hip_runtime.h>
#include <hip/hip_bf16.h>
#include <stdint.h>

#define TSEQ  2048
#define NH    16
#define NKV   4
#define HD    128
#define CDIM  2048
#define CKV   512
#define BATCH 2
#define MTOT  (BATCH * TSEQ)   // 4096 rows of (b,t)

typedef __attribute__((ext_vector_type(16))) __bf16 v16bf;
typedef __attribute__((ext_vector_type(8)))  __bf16 v8bf;
typedef __attribute__((ext_vector_type(8)))  float  v8f;

__device__ __forceinline__ unsigned short f2bf(float f) {
  union { float f; unsigned u; } v; v.f = f;
  unsigned u = v.u;
  u += 0x7FFFu + ((u >> 16) & 1u);   // round-to-nearest-even
  return (unsigned short)(u >> 16);
}
__device__ __forceinline__ float bf2f(unsigned short h) {
  union { unsigned u; float f; } v; v.u = ((unsigned)h) << 16;
  return v.f;
}

// Build a 16-half fragment from two 16-byte loads.
__device__ __forceinline__ v16bf load_frag(const __bf16* p0, const __bf16* p1) {
  v8bf lo = *reinterpret_cast<const v8bf*>(p0);
  v8bf hi = *reinterpret_cast<const v8bf*>(p1);
  v16bf r;
#pragma unroll
  for (int i = 0; i < 8; ++i) { r[i] = lo[i]; r[i + 8] = hi[i]; }
  return r;
}

__device__ __forceinline__ v8f vzero8() {
  v8f z;
#pragma unroll
  for (int i = 0; i < 8; ++i) z[i] = 0.0f;
  return z;
}

// ---------------- f32 -> bf16 convert ----------------
__global__ void k_convert_bf16(const float* __restrict__ in,
                               unsigned short* __restrict__ out, int n) {
  int stride = gridDim.x * blockDim.x;
  for (int i = blockIdx.x * blockDim.x + threadIdx.x; i < n; i += stride)
    out[i] = f2bf(in[i]);
}

// ---------------- WMMA GEMM: out[m,n] = sum_k A[m,k]*W[n,k] + bias[n] ----------------
// One wave computes a 64(M) x 32(N) tile. Ping/pong double-buffered k-loop
// (k-step 64 = two 32-k halves), K must be a multiple of 64.
__global__ void k_gemm_wmma(const unsigned short* __restrict__ Ag,
                            const unsigned short* __restrict__ Wg,
                            const float* __restrict__ bias,
                            void* __restrict__ out,
                            int M, int N, int K, int out_is_bf16) {
  const int lane = threadIdx.x & 31;
  const int l16  = lane & 15;
  const int hi   = lane >> 4;
  const int wave = (blockIdx.x * blockDim.x + threadIdx.x) >> 5;
  const int tiles_n = N >> 5;            // 32-column tiles
  const int tn = wave % tiles_n;
  const int tm = wave / tiles_n;
  const int mb = tm * 64;
  const int nb = tn * 32;
  if (mb >= M) return;

  const __bf16* A = reinterpret_cast<const __bf16*>(Ag);
  const __bf16* W = reinterpret_cast<const __bf16*>(Wg);

  // Per-lane base pointers (k-offset added in loop).
  const __bf16* aptr[4];
#pragma unroll
  for (int s = 0; s < 4; ++s)
    aptr[s] = A + (size_t)(mb + s * 16 + l16) * K + 8 * hi;   // A frag halves: +0 and +16
  const __bf16* wptr[2];
#pragma unroll
  for (int n = 0; n < 2; ++n)
    wptr[n] = W + (size_t)(nb + n * 16 + l16) * K + 8 * hi;  // B frag halves: +0 and +16

  v8f acc[4][2];
#pragma unroll
  for (int s = 0; s < 4; ++s)
#pragma unroll
    for (int n = 0; n < 2; ++n) acc[s][n] = vzero8();

  // Prologue: buffer 0 holds fragments for k = 0.
  v16bf af0[4], bf0[2], af1[4], bf1[2];
#pragma unroll
  for (int n = 0; n < 2; ++n) bf0[n] = load_frag(wptr[n], wptr[n] + 16);
#pragma unroll
  for (int s = 0; s < 4; ++s) af0[s] = load_frag(aptr[s], aptr[s] + 16);

  for (int k0 = 0; k0 < K; k0 += 64) {
    // --- half 1: load buffer 1 @ k0+32, compute on buffer 0 @ k0 ---
    const int k1 = k0 + 32;                       // always < K (K % 64 == 0)
#pragma unroll
    for (int n = 0; n < 2; ++n) {
      __builtin_prefetch(wptr[n] + k1 + 256, 0, 1);
      bf1[n] = load_frag(wptr[n] + k1, wptr[n] + k1 + 16);
    }
#pragma unroll
    for (int s = 0; s < 4; ++s) af1[s] = load_frag(aptr[s] + k1, aptr[s] + k1 + 16);
#pragma unroll
    for (int s = 0; s < 4; ++s)
#pragma unroll
      for (int n = 0; n < 2; ++n)
        acc[s][n] = __builtin_amdgcn_wmma_f32_16x16x32_bf16(
            false, af0[s], false, bf0[n], (short)0, acc[s][n], false, false);

    // --- half 2: load buffer 0 @ k0+64 (wrap on last), compute on buffer 1 ---
    const int k2 = (k0 + 64 < K) ? (k0 + 64) : 0;
#pragma unroll
    for (int n = 0; n < 2; ++n) bf0[n] = load_frag(wptr[n] + k2, wptr[n] + k2 + 16);
#pragma unroll
    for (int s = 0; s < 4; ++s) af0[s] = load_frag(aptr[s] + k2, aptr[s] + k2 + 16);
#pragma unroll
    for (int s = 0; s < 4; ++s)
#pragma unroll
      for (int n = 0; n < 2; ++n)
        acc[s][n] = __builtin_amdgcn_wmma_f32_16x16x32_bf16(
            false, af1[s], false, bf1[n], (short)0, acc[s][n], false, false);
  }

  if (out_is_bf16) {
    unsigned short* O = (unsigned short*)out;
#pragma unroll
    for (int n = 0; n < 2; ++n) {
      const int col = nb + n * 16 + l16;
      const float bv = bias ? bias[col] : 0.0f;
#pragma unroll
      for (int s = 0; s < 4; ++s)
#pragma unroll
        for (int r = 0; r < 8; ++r) {
          const int row = mb + s * 16 + r + 8 * hi;   // C/D layout: VGPR r, half hi
          O[(size_t)row * N + col] = f2bf(acc[s][n][r] + bv);
        }
    }
  } else {
    float* O = (float*)out;
#pragma unroll
    for (int n = 0; n < 2; ++n) {
      const int col = nb + n * 16 + l16;
      const float bv = bias ? bias[col] : 0.0f;
#pragma unroll
      for (int s = 0; s < 4; ++s)
#pragma unroll
        for (int r = 0; r < 8; ++r) {
          const int row = mb + s * 16 + r + 8 * hi;
          O[(size_t)row * N + col] = acc[s][n][r] + bv;
        }
    }
  }
}

// ---------------- RoPE in place on bf16 (B*T rows of nheads*128) ----------------
__global__ void k_rope(unsigned short* __restrict__ X, const float* __restrict__ cs,
                       const float* __restrict__ sn, int nheads, int n) {
  int stride = gridDim.x * blockDim.x;
  const int rowlen = nheads * HD;
  for (int idx = blockIdx.x * blockDim.x + threadIdx.x; idx < n; idx += stride) {
    const int i  = idx & 63;
    const int hh = (idx >> 6) % nheads;
    const int bt = idx / (64 * nheads);
    const int t  = bt % TSEQ;
    const size_t base = (size_t)bt * rowlen + hh * HD;
    const float x1 = bf2f(X[base + i]);
    const float x2 = bf2f(X[base + i + 64]);
    const float c = cs[t * 64 + i];
    const float s = sn[t * 64 + i];
    X[base + i]      = f2bf(x1 * c - x2 * s);
    X[base + i + 64] = f2bf(x1 * s + x2 * c);
  }
}

// ---------------- V transpose: V(b,t,g,d) -> VT(b,g,d,t) ----------------
__global__ void k_transpose_v(const unsigned short* __restrict__ V,
                              unsigned short* __restrict__ VT, int n) {
  int stride = gridDim.x * blockDim.x;
  for (int idx = blockIdx.x * blockDim.x + threadIdx.x; idx < n; idx += stride) {
    const int t = idx & 2047;
    const int d = (idx >> 11) & 127;
    const int g = (idx >> 18) & 3;
    const int b = idx >> 20;
    VT[idx] = V[(size_t)(b * TSEQ + t) * CKV + g * HD + d];
  }
}

// ---------------- Flash attention: one wave per (b, head, 16-query tile) ----------------
__global__ void k_attn(const unsigned short* __restrict__ Qg,
                       const unsigned short* __restrict__ Kg,
                       const unsigned short* __restrict__ VTg,
                       unsigned short* __restrict__ Yg) {
  __shared__ __align__(16) unsigned short ldsP[8][16 * 32];  // per-wave P tile (16x32 bf16)
  const int lane = threadIdx.x & 31;
  const int l16  = lane & 15;
  const int hi   = lane >> 4;
  const int w    = threadIdx.x >> 5;
  const int gw   = blockIdx.x * (blockDim.x >> 5) + w;
  const int qt = gw & 127;          // T/16 = 128 query tiles
  const int h  = (gw >> 7) & 15;
  const int b  = gw >> 11;
  const int g  = h >> 2;            // GQA group (N_REP = 4)
  const int m0 = qt * 16;

  const __bf16* Q  = reinterpret_cast<const __bf16*>(Qg);
  const __bf16* K  = reinterpret_cast<const __bf16*>(Kg);
  const __bf16* VT = reinterpret_cast<const __bf16*>(VTg);

  // Resident Q fragments: 16x128 as 4 chunks of K=32.
  v16bf qf[4];
#pragma unroll
  for (int c = 0; c < 4; ++c) {
    const __bf16* qp = Q + (size_t)(b * TSEQ + m0 + l16) * CDIM + h * HD + c * 32 + 16 * hi;
    qf[c] = load_frag(qp, qp + 8);
  }

  v8f o[8];
#pragma unroll
  for (int t = 0; t < 8; ++t) o[t] = vzero8();
  float mrow[8], lrow[8];
#pragma unroll
  for (int r = 0; r < 8; ++r) { mrow[r] = -1e30f; lrow[r] = 0.0f; }

  const float scale = 0.08838834764831845f;   // 1/sqrt(128)
  const float LOG2E = 1.4426950408889634f;
  const int nblocks = (m0 + 47) >> 5;         // key blocks of 32, up to row m0+15

  for (int jb = 0; jb < nblocks; ++jb) {
    const int kb0 = jb * 32;
    // Batch all 8 K^T fragment loads, then the 8 QK WMMAs.
    v16bf kf[8];
#pragma unroll
    for (int c = 0; c < 4; ++c) {
      const __bf16* kp1 = K + (size_t)(b * TSEQ + kb0 + l16) * CKV + g * HD + c * 32 + 16 * hi;
      kf[c] = load_frag(kp1, kp1 + 8);
      const __bf16* kp2 = K + (size_t)(b * TSEQ + kb0 + 16 + l16) * CKV + g * HD + c * 32 + 16 * hi;
      kf[4 + c] = load_frag(kp2, kp2 + 8);
    }
    v8f s1 = vzero8(), s2 = vzero8();
#pragma unroll
    for (int c = 0; c < 4; ++c)
      s1 = __builtin_amdgcn_wmma_f32_16x16x32_bf16(false, qf[c], false, kf[c], (short)0, s1, false, false);
#pragma unroll
    for (int c = 0; c < 4; ++c)
      s2 = __builtin_amdgcn_wmma_f32_16x16x32_bf16(false, qf[c], false, kf[4 + c], (short)0, s2, false, false);

    // Online softmax over the 32-key block.
#pragma unroll
    for (int r = 0; r < 8; ++r) {
      const int row = m0 + r + 8 * hi;
      float a  = (kb0 + l16      <= row) ? s1[r] * scale : -1e30f;
      float bb = (kb0 + 16 + l16 <= row) ? s2[r] * scale : -1e30f;
      float mx = fmaxf(a, bb);
      mx = fmaxf(mx, __shfl_xor(mx, 1, 32));
      mx = fmaxf(mx, __shfl_xor(mx, 2, 32));
      mx = fmaxf(mx, __shfl_xor(mx, 4, 32));
      mx = fmaxf(mx, __shfl_xor(mx, 8, 32));
      const float mnew  = fmaxf(mrow[r], mx);
      const float alpha = exp2f((mrow[r] - mnew) * LOG2E);
      const float p1 = exp2f((a - mnew) * LOG2E);
      const float p2 = exp2f((bb - mnew) * LOG2E);
      float ps = p1 + p2;
      ps += __shfl_xor(ps, 1, 32);
      ps += __shfl_xor(ps, 2, 32);
      ps += __shfl_xor(ps, 4, 32);
      ps += __shfl_xor(ps, 8, 32);
      lrow[r] = lrow[r] * alpha + ps;
      mrow[r] = mnew;
#pragma unroll
      for (int t = 0; t < 8; ++t) o[t][r] *= alpha;
      // C-layout -> LDS (row-major 16x32 bf16) for A-layout reload.
      ldsP[w][(r + 8 * hi) * 32 + l16]      = f2bf(p1);
      ldsP[w][(r + 8 * hi) * 32 + 16 + l16] = f2bf(p2);
    }
    asm volatile("s_wait_dscnt 0" ::: "memory");  // wave-local LDS transpose hand-off
    const __bf16* pp = reinterpret_cast<const __bf16*>(&ldsP[w][l16 * 32 + 16 * hi]);
    v16bf pf = load_frag(pp, pp + 8);            // P as A-fragment (16x32)

    // Batch all 8 V fragment loads, then the 8 PV WMMAs.
    v16bf vf[8];
#pragma unroll
    for (int t = 0; t < 8; ++t) {
      const __bf16* vp = VT + ((size_t)(b * NKV + g) * HD + t * 16 + l16) * TSEQ + kb0 + 16 * hi;
      vf[t] = load_frag(vp, vp + 8);
    }
#pragma unroll
    for (int t = 0; t < 8; ++t)
      o[t] = __builtin_amdgcn_wmma_f32_16x16x32_bf16(false, pf, false, vf[t], (short)0, o[t], false, false);
  }

#pragma unroll
  for (int r = 0; r < 8; ++r) {
    const float inv = 1.0f / lrow[r];
    const int row = m0 + r + 8 * hi;
#pragma unroll
    for (int t = 0; t < 8; ++t)
      Yg[(size_t)(b * TSEQ + row) * CDIM + h * HD + t * 16 + l16] = f2bf(o[t][r] * inv);
  }
}

// ---------------- host-side launch ----------------
extern "C" void kernel_launch(void* const* d_in, const int* in_sizes, int n_in,
                              void* d_out, int out_size, void* d_ws, size_t ws_size,
                              hipStream_t stream) {
  (void)in_sizes; (void)n_in; (void)out_size; (void)ws_size;
  const float* x    = (const float*)d_in[0];
  const float* cosp = (const float*)d_in[1];
  const float* sinp = (const float*)d_in[2];
  const float* wq   = (const float*)d_in[3];
  const float* bq   = (const float*)d_in[4];
  const float* wk   = (const float*)d_in[5];
  const float* bk   = (const float*)d_in[6];
  const float* wv   = (const float*)d_in[7];
  const float* bv   = (const float*)d_in[8];
  const float* wp   = (const float*)d_in[9];
  float* out = (float*)d_out;

  char* ws = (char*)d_ws;
  size_t off = 0;
  auto take = [&](size_t elems) {
    unsigned short* p = (unsigned short*)(ws + off);
    off += (elems * sizeof(unsigned short) + 255) & ~(size_t)255;
    return p;
  };
  unsigned short* xb  = take((size_t)MTOT * CDIM);
  unsigned short* wqb = take((size_t)CDIM * CDIM);
  unsigned short* wkb = take((size_t)CKV * CDIM);
  unsigned short* wvb = take((size_t)CKV * CDIM);
  unsigned short* wpb = take((size_t)CDIM * CDIM);
  unsigned short* qb  = take((size_t)MTOT * CDIM);
  unsigned short* kb  = take((size_t)MTOT * CKV);
  unsigned short* vb  = take((size_t)MTOT * CKV);
  unsigned short* vtb = take((size_t)MTOT * CKV);
  unsigned short* yb  = take((size_t)MTOT * CDIM);

  dim3 blk(256);
  auto cvt = [&](const float* src, unsigned short* dst, int n) {
    int blocks = (n + 256 * 8 - 1) / (256 * 8);
    k_convert_bf16<<<blocks, blk, 0, stream>>>(src, dst, n);
  };
  cvt(x,  xb,  MTOT * CDIM);
  cvt(wq, wqb, CDIM * CDIM);
  cvt(wk, wkb, CKV * CDIM);
  cvt(wv, wvb, CKV * CDIM);
  cvt(wp, wpb, CDIM * CDIM);

  auto gemm = [&](const unsigned short* A, const unsigned short* W, const float* bias,
                  void* O, int M, int N, int K, int obf) {
    int waves = (M / 64) * (N / 32);
    int blocks = (waves * 32 + 255) / 256;
    k_gemm_wmma<<<blocks, blk, 0, stream>>>(A, W, bias, O, M, N, K, obf);
  };
  gemm(xb, wqb, bq, qb, MTOT, CDIM, CDIM, 1);
  gemm(xb, wkb, bk, kb, MTOT, CKV,  CDIM, 1);
  gemm(xb, wvb, bv, vb, MTOT, CKV,  CDIM, 1);

  {
    int nq = MTOT * NH * 64;
    k_rope<<<(nq + 2047) / 2048, blk, 0, stream>>>(qb, cosp, sinp, NH, nq);
    int nk = MTOT * NKV * 64;
    k_rope<<<(nk + 2047) / 2048, blk, 0, stream>>>(kb, cosp, sinp, NKV, nk);
  }
  {
    int nt = MTOT * CKV;
    k_transpose_v<<<(nt + 2047) / 2048, blk, 0, stream>>>(vb, vtb, nt);
  }
  {
    int waves = BATCH * NH * (TSEQ / 16);   // 4096 waves
    int blocks = waves * 32 / 256;          // 512 blocks
    k_attn<<<blocks, blk, 0, stream>>>(qb, kb, vtb, yb);
  }
  gemm(yb, wpb, nullptr, out, MTOT, CDIM, CDIM, 0);
}
